// CausalSelfAttention_33655363732189
// MI455X (gfx1250) — matrix operand order repaired and checked by reading, not compiled, not run on hardware
//
#include <hip/hip_runtime.h>
#include <math.h>

// Problem constants
#define BB 4
#define TT 2048
#define DD 1024
#define HH 16
#define HD 64
#define MTOT (BB*TT)          // 8192 tokens

typedef __attribute__((ext_vector_type(16))) __bf16 v16bf;
typedef __attribute__((ext_vector_type(8)))  __bf16 v8bf;
typedef __attribute__((ext_vector_type(8)))  float  v8f;

union BF16V { v16bf v; v8bf h[2]; };

__device__ __forceinline__ unsigned short f2bf(float f) {
  unsigned u = __float_as_uint(f);
  u += 0x7FFFu + ((u >> 16) & 1u);   // round-to-nearest-even
  return (unsigned short)(u >> 16);
}

// B-operand (32x16 bf16): 16 contiguous bf16 per lane
__device__ __forceinline__ v16bf ld_b16(const unsigned short* p) {
  BF16V u;
  const v8bf* q = reinterpret_cast<const v8bf*>(p);
  u.h[0] = q[0]; u.h[1] = q[1];
  return u.v;
}

// A-operand (16x32 bf16) from a row-major row: chunks [k+16h,+8) and [k+16h+8,+8)
__device__ __forceinline__ v16bf ld_a16(const unsigned short* row, int k, int half) {
  BF16V u;
  u.h[0] = *reinterpret_cast<const v8bf*>(row + k + 16*half);
  u.h[1] = *reinterpret_cast<const v8bf*>(row + k + 16*half + 8);
  return u.v;
}

__device__ __forceinline__ v8f wmma_bf16(v16bf a, v16bf b, v8f c) {
  return __builtin_amdgcn_wmma_f32_16x16x32_bf16(false, a, false, b, (short)0, c,
                                                 false, false);
}

#define V8F_ZERO {0.f,0.f,0.f,0.f,0.f,0.f,0.f,0.f}

// ---------------- fp32 -> bf16 converts ----------------
__global__ void cvt_f32_bf16(const float* __restrict__ in,
                             unsigned short* __restrict__ out, int n) {
  int i = blockIdx.x * blockDim.x + threadIdx.x;
  if (i < n) out[i] = f2bf(in[i]);
}

// out[n*K + k] = in[k*N + n]   (weight transpose to N-major for B operands)
__global__ void cvt_transpose_bf16(const float* __restrict__ in,
                                   unsigned short* __restrict__ out, int N, int K) {
  int i = blockIdx.x * blockDim.x + threadIdx.x;
  if (i >= N * K) return;
  int n = i / K, k = i - n * K;
  out[i] = f2bf(in[k * N + n]);
}

// ---------------- QKV GEMM: [8192,1024] x [1024,3072] -> scatter Q/K/Vt ----
__global__ __launch_bounds__(256) void qkv_gemm(
    const unsigned short* __restrict__ xb,   // [8192][1024] bf16
    const unsigned short* __restrict__ wT,   // [3072][1024] bf16 (N-major)
    const float* __restrict__ bias,          // [3072]
    unsigned short* __restrict__ Qb,         // [B,H,T,HD] (scaled)
    unsigned short* __restrict__ Kb,         // [B,H,T,HD]
    unsigned short* __restrict__ Vt)         // [B,H,HD,T]
{
  const int lane = threadIdx.x & 31;
  const int half = lane >> 4;
  const int l16  = lane & 15;
  const int wg   = blockIdx.x * 8 + (threadIdx.x >> 5);
  const int NT   = (3 * DD) / 64;            // 48 column tiles
  const int m0   = (wg / NT) * 32;
  const int n0   = (wg % NT) * 64;

  v8f acc[2][4];
  #pragma unroll
  for (int a = 0; a < 2; a++)
    #pragma unroll
    for (int j = 0; j < 4; j++) { v8f z = V8F_ZERO; acc[a][j] = z; }

  const unsigned short* arow0 = xb + (m0 + l16) * DD;
  const unsigned short* arow1 = xb + (m0 + 16 + l16) * DD;
  const unsigned short* brow[4];
  #pragma unroll
  for (int j = 0; j < 4; j++) brow[j] = wT + (n0 + 16*j + l16) * DD;

  for (int k = 0; k < DD; k += 32) {
    v16bf a0 = ld_a16(arow0, k, half);
    v16bf a1 = ld_a16(arow1, k, half);
    #pragma unroll
    for (int j = 0; j < 4; j++) {
      v16bf b = ld_b16(brow[j] + k + 16*half);
      acc[0][j] = wmma_bf16(a0, b, acc[0][j]);
      acc[1][j] = wmma_bf16(a1, b, acc[1][j]);
    }
  }

  const int bcst = m0 >> 11;                 // batch index (32 | 2048)
  #pragma unroll
  for (int j = 0; j < 4; j++) {
    const int n  = n0 + 16*j + l16;
    const float bj = bias[n];
    const int nm = n & (DD - 1);
    const int h  = nm >> 6, hd = nm & 63;
    unsigned short* outp;
    int stride; float scl;
    if (n < DD)          { outp = Qb + ((bcst*HH + h)*TT)*HD + hd; stride = HD; scl = 0.125f; }
    else if (n < 2*DD)   { outp = Kb + ((bcst*HH + h)*TT)*HD + hd; stride = HD; scl = 1.0f;  }
    else                 { outp = Vt + ((bcst*HH + h)*HD + hd)*TT; stride = 1;  scl = 1.0f;  }
    #pragma unroll
    for (int ms = 0; ms < 2; ms++)
      #pragma unroll
      for (int r = 0; r < 8; r++) {
        int m = m0 + 16*ms + r + 8*half;
        int t = m & (TT - 1);
        outp[t * stride] = f2bf((acc[ms][j][r] + bj) * scl);
      }
  }
}

// ---------------- Flash attention: one wave = 16 queries x full HD ---------
__global__ __launch_bounds__(256) void attn_flash(
    const unsigned short* __restrict__ Qb,   // [B,H,T,HD] scaled
    const unsigned short* __restrict__ Kb,   // [B,H,T,HD]
    const unsigned short* __restrict__ Vt,   // [B,H,HD,T]
    unsigned short* __restrict__ Yb)         // [B,T,D] bf16
{
  __shared__ __align__(16) unsigned short Pls[8][16 * 40]; // per-wave 16x32, stride 40
  const int lane = threadIdx.x & 31;
  const int half = lane >> 4;
  const int l16  = lane & 15;
  const int wib  = threadIdx.x >> 5;
  const int wg   = blockIdx.x * 8 + wib;
  const int qt   = wg & ((TT/16) - 1);       // 128 q-tiles
  const int bh   = wg >> 7;
  const int q0   = qt * 16;
  unsigned short* P = &Pls[wib][0];

  const unsigned short* qrow = Qb + ((long)bh * TT + q0 + l16) * HD;
  const v16bf a0 = ld_a16(qrow, 0,  half);
  const v16bf a1 = ld_a16(qrow, 32, half);

  v8f acc[4];
  float mrow[8], srow[8];
  #pragma unroll
  for (int t = 0; t < 4; t++) { v8f z = V8F_ZERO; acc[t] = z; }
  #pragma unroll
  for (int r = 0; r < 8; r++) { mrow[r] = -INFINITY; srow[r] = 0.0f; }

  for (int kb = 0; kb < q0 + 16; kb += 32) {
    v8f s0 = V8F_ZERO, s1 = V8F_ZERO;
    {
      const unsigned short* k0 = Kb + ((long)bh * TT + kb + l16) * HD + 16*half;
      s0 = wmma_bf16(a0, ld_b16(k0),      s0);
      s0 = wmma_bf16(a1, ld_b16(k0 + 32), s0);
      const unsigned short* k1 = Kb + ((long)bh * TT + kb + 16 + l16) * HD + 16*half;
      s1 = wmma_bf16(a0, ld_b16(k1),      s1);
      s1 = wmma_bf16(a1, ld_b16(k1 + 32), s1);
    }
    const int key0 = kb + l16;
    const int key1 = kb + 16 + l16;
    #pragma unroll
    for (int r = 0; r < 8; r++) {
      const int q = q0 + r + 8*half;
      float v0 = (key0 <= q) ? s0[r] : -INFINITY;
      float v1 = (key1 <= q) ? s1[r] : -INFINITY;
      float mx = fmaxf(v0, v1);
      mx = fmaxf(mx, __shfl_xor(mx, 1));
      mx = fmaxf(mx, __shfl_xor(mx, 2));
      mx = fmaxf(mx, __shfl_xor(mx, 4));
      mx = fmaxf(mx, __shfl_xor(mx, 8));   // rows live within 16-lane halves
      const float mnew = fmaxf(mrow[r], mx);
      const float corr = __expf(mrow[r] - mnew);
      const float p0 = __expf(v0 - mnew);
      const float p1 = __expf(v1 - mnew);
      float rs = p0 + p1;
      rs += __shfl_xor(rs, 1); rs += __shfl_xor(rs, 2);
      rs += __shfl_xor(rs, 4); rs += __shfl_xor(rs, 8);
      srow[r] = srow[r] * corr + rs;
      mrow[r] = mnew;
      #pragma unroll
      for (int t = 0; t < 4; t++) acc[t][r] *= corr;
      const int row = r + 8*half;
      P[row * 40 + l16]      = f2bf(p0);
      P[row * 40 + 16 + l16] = f2bf(p1);
    }
    // P (16x32) back as A operand via LDS transpose (same-wave, in-order DS)
    BF16V pu;
    pu.h[0] = *reinterpret_cast<const v8bf*>(P + l16 * 40 + 16*half);
    pu.h[1] = *reinterpret_cast<const v8bf*>(P + l16 * 40 + 16*half + 8);
    #pragma unroll
    for (int t = 0; t < 4; t++) {
      const unsigned short* vp =
          Vt + ((long)bh * HD + 16*t + l16) * TT + kb + 16*half;
      acc[t] = wmma_bf16(pu.v, ld_b16(vp), acc[t]);
    }
  }

  const int b = bh >> 4, h = bh & 15;
  #pragma unroll
  for (int r = 0; r < 8; r++) {
    const float inv = 1.0f / srow[r];
    const int q = q0 + r + 8*half;
    unsigned short* yp = Yb + ((long)(b * TT + q)) * DD + h * HD + l16;
    #pragma unroll
    for (int t = 0; t < 4; t++) yp[16 * t] = f2bf(acc[t][r] * inv);
  }
}

// ---------------- Output projection: [8192,1024] x [1024,1024] + bias -> f32
__global__ __launch_bounds__(256) void proj_gemm(
    const unsigned short* __restrict__ yb,   // [8192][1024] bf16
    const unsigned short* __restrict__ wT,   // [1024][1024] bf16 (N-major)
    const float* __restrict__ bias,          // [1024]
    float* __restrict__ out)                 // [8192][1024] f32
{
  const int lane = threadIdx.x & 31;
  const int half = lane >> 4;
  const int l16  = lane & 15;
  const int wg   = blockIdx.x * 8 + (threadIdx.x >> 5);
  const int NT   = DD / 64;                  // 16
  const int m0   = (wg / NT) * 32;
  const int n0   = (wg % NT) * 64;

  v8f acc[2][4];
  #pragma unroll
  for (int a = 0; a < 2; a++)
    #pragma unroll
    for (int j = 0; j < 4; j++) { v8f z = V8F_ZERO; acc[a][j] = z; }

  const unsigned short* arow0 = yb + (m0 + l16) * DD;
  const unsigned short* arow1 = yb + (m0 + 16 + l16) * DD;
  const unsigned short* brow[4];
  #pragma unroll
  for (int j = 0; j < 4; j++) brow[j] = wT + (n0 + 16*j + l16) * DD;

  for (int k = 0; k < DD; k += 32) {
    v16bf a0 = ld_a16(arow0, k, half);
    v16bf a1 = ld_a16(arow1, k, half);
    #pragma unroll
    for (int j = 0; j < 4; j++) {
      v16bf b = ld_b16(brow[j] + k + 16*half);
      acc[0][j] = wmma_bf16(a0, b, acc[0][j]);
      acc[1][j] = wmma_bf16(a1, b, acc[1][j]);
    }
  }

  #pragma unroll
  for (int j = 0; j < 4; j++) {
    const int n = n0 + 16*j + l16;
    const float bj = bias[n];
    #pragma unroll
    for (int ms = 0; ms < 2; ms++)
      #pragma unroll
      for (int r = 0; r < 8; r++) {
        const int m = m0 + 16*ms + r + 8*half;
        out[(long)m * DD + n] = acc[ms][j][r] + bj;
      }
  }
}

extern "C" void kernel_launch(void* const* d_in, const int* in_sizes, int n_in,
                              void* d_out, int out_size, void* d_ws, size_t ws_size,
                              hipStream_t stream) {
  const float* x      = (const float*)d_in[0];   // [B,T,D]
  const float* w_qkv  = (const float*)d_in[1];   // [D,3D]
  const float* b_qkv  = (const float*)d_in[2];   // [3D]
  const float* w_proj = (const float*)d_in[3];   // [D,D]
  const float* b_proj = (const float*)d_in[4];   // [D]
  float* out = (float*)d_out;

  char* ws = (char*)d_ws;
  const size_t SZ_X   = (size_t)MTOT * DD * 2;        // 16.78 MB
  const size_t SZ_WQ  = (size_t)3 * DD * DD * 2;      //  6.29 MB
  const size_t SZ_WP  = (size_t)DD * DD * 2;          //  2.10 MB
  const size_t SZ_QKV = (size_t)BB * HH * TT * HD * 2;// 16.78 MB each
  unsigned short* xb     = (unsigned short*)(ws);
  unsigned short* wqkvT  = (unsigned short*)(ws + SZ_X);
  unsigned short* wprojT = (unsigned short*)(ws + SZ_X + SZ_WQ);
  unsigned short* Qb     = (unsigned short*)(ws + SZ_X + SZ_WQ + SZ_WP);
  unsigned short* Kb     = (unsigned short*)(ws + SZ_X + SZ_WQ + SZ_WP + SZ_QKV);
  unsigned short* Vt     = (unsigned short*)(ws + SZ_X + SZ_WQ + SZ_WP + 2*SZ_QKV);
  unsigned short* Yb     = (unsigned short*)(ws + SZ_X + SZ_WQ + SZ_WP + 3*SZ_QKV);

  const int nx = MTOT * DD;                 // 8,388,608
  cvt_f32_bf16<<<(nx + 255) / 256, 256, 0, stream>>>(x, xb, nx);
  cvt_transpose_bf16<<<(3 * DD * DD + 255) / 256, 256, 0, stream>>>(w_qkv, wqkvT, 3 * DD, DD);
  cvt_transpose_bf16<<<(DD * DD + 255) / 256, 256, 0, stream>>>(w_proj, wprojT, DD, DD);

  // (M/32)*(3D/64) waves / 8 per block = 1536 blocks
  qkv_gemm<<<(MTOT / 32) * (3 * DD / 64) / 8, 256, 0, stream>>>(
      xb, wqkvT, b_qkv, Qb, Kb, Vt);

  // B*H*(T/16) waves / 8 = 1024 blocks
  attn_flash<<<BB * HH * (TT / 16) / 8, 256, 0, stream>>>(Qb, Kb, Vt, Yb);

  // (M/32)*(D/64)/8 = 512 blocks
  proj_gemm<<<(MTOT / 32) * (DD / 64) / 8, 256, 0, stream>>>(Yb, wprojT, b_proj, out);
}
